// GraphTransformerEdgeEncoder_69312182223529
// MI455X (gfx1250) — compile-verified, matrix operand order checked by reading, not run on hardware
//
#include <hip/hip_runtime.h>
#include <hip/hip_bf16.h>
#include <math.h>

typedef __bf16 bf16_t;
typedef __attribute__((ext_vector_type(16))) __bf16 v16bf;
typedef __attribute__((ext_vector_type(8)))  float  v8f;

#define WMMA_BF16(a, b, c) \
  __builtin_amdgcn_wmma_f32_16x16x32_bf16(false, (a), false, (b), (short)0, (c), false, false)

static constexpr int E_EDGES = 8192;
static constexpr int NNODES  = 2048;
static constexpr int HID     = 128;
static constexpr int HEADS   = 4;
static constexpr int DH      = 32;

// ---------------------------------------------------------------------------
// Tiled GEMM: C[M,N] = A[M,K] @ B (+bias) (+residual) (+relu).
// One wave per 16x64 output tile (4 WMMAs per K-step, A reused).
// LDS tiles are stored K-contiguous per operand row/col so every WMMA operand
// gather is two ds_load_b128 per lane.
// TRANSB=false: B is [K,N] row-major. TRANSB=true: B is [N,K] row-major (B^T).
// Requires: M % 16 == 0, N % 64 == 0, K % 32 == 0, 16B-aligned pointers.
// ---------------------------------------------------------------------------
template<bool TRANSB, bool RELU, bool RESID>
__global__ void gemm_bf16_wmma(const float* __restrict__ A, const float* __restrict__ B,
                               const float* __restrict__ bias, const float* __restrict__ R,
                               float* __restrict__ C, int M, int Nn, int K) {
  __shared__ __bf16 sA[16 * 32];   // [row][k]
  __shared__ __bf16 sB[64 * 32];   // [n][k]  (B^T tile)
  const int ntn  = Nn >> 6;
  const int tm   = blockIdx.x / ntn;
  const int tn   = blockIdx.x % ntn;
  const int m0   = tm << 4, n0 = tn << 6;
  const int lane = threadIdx.x;
  const int half = lane >> 4;
  const int rl   = lane & 15;
  const int kb   = half << 3;           // 0 or 8

  v8f acc[4] = {v8f{}, v8f{}, v8f{}, v8f{}};
  for (int k0 = 0; k0 < K; k0 += 32) {
    // stage A tile (16 rows x 32 k), float4 global loads, K-contiguous in LDS
    for (int i = lane; i < 128; i += 32) {
      int r = i >> 3, c4 = (i & 7) << 2;
      float4 vv = *reinterpret_cast<const float4*>(A + (size_t)(m0 + r) * K + k0 + c4);
      sA[r * 32 + c4 + 0] = (__bf16)vv.x;
      sA[r * 32 + c4 + 1] = (__bf16)vv.y;
      sA[r * 32 + c4 + 2] = (__bf16)vv.z;
      sA[r * 32 + c4 + 3] = (__bf16)vv.w;
    }
    // stage B^T tile (64 n x 32 k)
    if (TRANSB) {
      for (int i = lane; i < 512; i += 32) {       // 64 n x 8 k-quads
        int n = i >> 3, kg = (i & 7) << 2;
        float4 vv = *reinterpret_cast<const float4*>(B + (size_t)(n0 + n) * K + k0 + kg);
        sB[n * 32 + kg + 0] = (__bf16)vv.x;
        sB[n * 32 + kg + 1] = (__bf16)vv.y;
        sB[n * 32 + kg + 2] = (__bf16)vv.z;
        sB[n * 32 + kg + 3] = (__bf16)vv.w;
      }
    } else {
      for (int i = lane; i < 512; i += 32) {       // 32 k x 16 n-quads
        int kk = i >> 4, ng = (i & 15) << 2;
        float4 vv = *reinterpret_cast<const float4*>(B + (size_t)(k0 + kk) * Nn + n0 + ng);
        sB[(ng + 0) * 32 + kk] = (__bf16)vv.x;
        sB[(ng + 1) * 32 + kk] = (__bf16)vv.y;
        sB[(ng + 2) * 32 + kk] = (__bf16)vv.z;
        sB[(ng + 3) * 32 + kk] = (__bf16)vv.w;
      }
    }
    __syncthreads();
    v16bf a;
#pragma unroll
    for (int i = 0; i < 8; ++i) {
      a[i]     = sA[rl * 32 + kb + i];
      a[8 + i] = sA[rl * 32 + kb + 16 + i];
    }
#pragma unroll
    for (int t = 0; t < 4; ++t) {
      const int nb = t << 4;
      v16bf b;
#pragma unroll
      for (int i = 0; i < 8; ++i) {
        b[i]     = sB[(nb + rl) * 32 + kb + i];
        b[8 + i] = sB[(nb + rl) * 32 + kb + 16 + i];
      }
      acc[t] = WMMA_BF16(a, b, acc[t]);
    }
    __syncthreads();
  }
  const int rbase = half << 3;
#pragma unroll
  for (int t = 0; t < 4; ++t) {
#pragma unroll
    for (int j = 0; j < 8; ++j) {
      int mm2 = m0 + j + rbase;
      int nn  = n0 + (t << 4) + rl;
      float v = acc[t][j] + bias[nn];
      if (RESID) v += R[(size_t)mm2 * Nn + nn];
      if (RELU)  v = fmaxf(v, 0.0f);
      C[(size_t)mm2 * Nn + nn] = v;
    }
  }
}

// ---------------------------------------------------------------------------
// Flash attention: one wave per (16 query rows, head). Streams 32-key K/V
// tiles through LDS with online softmax. V is staged transposed ([dim][key])
// so its WMMA B gather is K-contiguous (ds_load_b128).
// ---------------------------------------------------------------------------
__global__ void flash_attn_kernel(const float* __restrict__ Q, const float* __restrict__ K,
                                  const float* __restrict__ V, float* __restrict__ O) {
  __shared__ __bf16 sK[32 * 32];   // [key][dim]
  __shared__ __bf16 sVt[32 * 32];  // [dim][key]
  __shared__ __bf16 sP[16 * 32];   // [qrow][key]
  const int head = blockIdx.y;
  const int q0   = blockIdx.x << 4;
  const int lane = threadIdx.x;
  const int half = lane >> 4;
  const int rl   = lane & 15;
  const int kb   = half << 3;
  const float scale = 0.17677669529663687f;  // 1/sqrt(32)

  // Q A-tile, held in registers for the whole sweep
  v16bf aq;
  {
    const float* qr = Q + (size_t)(q0 + rl) * HID + head * DH;
#pragma unroll
    for (int i = 0; i < 8; ++i) {
      aq[i]     = (__bf16)qr[kb + i];
      aq[8 + i] = (__bf16)qr[kb + 16 + i];
    }
  }
  float m[8], l[8];
  v8f acc0 = {}, acc1 = {};
#pragma unroll
  for (int j = 0; j < 8; ++j) { m[j] = -1e30f; l[j] = 0.0f; }

  for (int k0 = 0; k0 < E_EDGES; k0 += 32) {
    // stage K [key][dim] and V^T [dim][key], float4 global loads
    for (int i = lane; i < 256; i += 32) {
      int key = i >> 3, dg = (i & 7) << 2;
      size_t g = (size_t)(k0 + key) * HID + head * DH + dg;
      float4 kv = *reinterpret_cast<const float4*>(K + g);
      float4 vv = *reinterpret_cast<const float4*>(V + g);
      sK[key * 32 + dg + 0] = (__bf16)kv.x;
      sK[key * 32 + dg + 1] = (__bf16)kv.y;
      sK[key * 32 + dg + 2] = (__bf16)kv.z;
      sK[key * 32 + dg + 3] = (__bf16)kv.w;
      sVt[(dg + 0) * 32 + key] = (__bf16)vv.x;
      sVt[(dg + 1) * 32 + key] = (__bf16)vv.y;
      sVt[(dg + 2) * 32 + key] = (__bf16)vv.z;
      sVt[(dg + 3) * 32 + key] = (__bf16)vv.w;
    }
    if (k0 + 32 < E_EDGES)  // hint next K tile toward the caches
      __builtin_prefetch(K + (size_t)(k0 + 32 + (lane & 31)) * HID + head * DH, 0, 1);
    __syncthreads();

    // S tiles: B[k=dim][n=key] gathered K-contiguous from sK[key][dim]
    v16bf b0, b1;
#pragma unroll
    for (int i = 0; i < 8; ++i) {
      b0[i]     = sK[rl * 32 + kb + i];
      b0[8 + i] = sK[rl * 32 + kb + 16 + i];
      b1[i]     = sK[(16 + rl) * 32 + kb + i];
      b1[8 + i] = sK[(16 + rl) * 32 + kb + 16 + i];
    }
    v8f zero = {};
    v8f s0 = WMMA_BF16(aq, b0, zero);
    v8f s1 = WMMA_BF16(aq, b1, zero);

#pragma unroll
    for (int j = 0; j < 8; ++j) {
      float v0 = s0[j] * scale, v1 = s1[j] * scale;
      float rm = fmaxf(v0, v1);
      rm = fmaxf(rm, __shfl_xor(rm, 1, 32));
      rm = fmaxf(rm, __shfl_xor(rm, 2, 32));
      rm = fmaxf(rm, __shfl_xor(rm, 4, 32));
      rm = fmaxf(rm, __shfl_xor(rm, 8, 32));
      float mn   = fmaxf(m[j], rm);
      float corr = __expf(m[j] - mn);
      float p0 = __expf(v0 - mn), p1 = __expf(v1 - mn);
      float rs = p0 + p1;
      rs += __shfl_xor(rs, 1, 32);
      rs += __shfl_xor(rs, 2, 32);
      rs += __shfl_xor(rs, 4, 32);
      rs += __shfl_xor(rs, 8, 32);
      l[j] = l[j] * corr + rs;
      m[j] = mn;
      acc0[j] *= corr;
      acc1[j] *= corr;
      int row = j + (half << 3);           // C-layout row of element j
      sP[row * 32 + rl]      = (__bf16)p0;
      sP[row * 32 + 16 + rl] = (__bf16)p1;
    }
    __syncthreads();

    // P as A (16 q-rows x 32 keys); V as B[k=key][n=dim] from sVt[dim][key]
    v16bf ap, bv0, bv1;
#pragma unroll
    for (int i = 0; i < 8; ++i) {
      ap[i]      = sP[rl * 32 + kb + i];
      ap[8 + i]  = sP[rl * 32 + kb + 16 + i];
      bv0[i]     = sVt[rl * 32 + kb + i];
      bv0[8 + i] = sVt[rl * 32 + kb + 16 + i];
      bv1[i]     = sVt[(16 + rl) * 32 + kb + i];
      bv1[8 + i] = sVt[(16 + rl) * 32 + kb + 16 + i];
    }
    acc0 = WMMA_BF16(ap, bv0, acc0);
    acc1 = WMMA_BF16(ap, bv1, acc1);
    __syncthreads();
  }
#pragma unroll
  for (int j = 0; j < 8; ++j) {
    int row = q0 + j + (half << 3);
    float inv = 1.0f / l[j];
    O[(size_t)row * HID + head * DH + rl]      = acc0[j] * inv;
    O[(size_t)row * HID + head * DH + 16 + rl] = acc1[j] * inv;
  }
}

// ---------------------------------------------------------------------------
// LayerNorm over 128 cols; input already contains residual sum.
// ---------------------------------------------------------------------------
__global__ void ln_kernel(const float* __restrict__ X, const float* __restrict__ g,
                          const float* __restrict__ b, float* __restrict__ Out) {
  __shared__ float red[128];
  const int e = blockIdx.x, c = threadIdx.x;
  float v = X[(size_t)e * HID + c];
  red[c] = v;
  __syncthreads();
  for (int s = 64; s > 0; s >>= 1) { if (c < s) red[c] += red[c + s]; __syncthreads(); }
  float mean = red[0] * (1.0f / 128.0f);
  __syncthreads();
  float d = v - mean;
  red[c] = d * d;
  __syncthreads();
  for (int s = 64; s > 0; s >>= 1) { if (c < s) red[c] += red[c + s]; __syncthreads(); }
  float var = red[0] * (1.0f / 128.0f);
  Out[(size_t)e * HID + c] = d * rsqrtf(var + 1e-5f) * g[c] + b[c];
}

// ---------------------------------------------------------------------------
// timestamp min/max (single block)
// ---------------------------------------------------------------------------
__global__ void minmax_kernel(const float* __restrict__ ts, float* __restrict__ mm) {
  __shared__ float smin[256], smax[256];
  const int t = threadIdx.x;
  float mn = 1e30f, mx = -1e30f;
  for (int i = t; i < E_EDGES; i += 256) {
    float v = ts[i];
    mn = fminf(mn, v);
    mx = fmaxf(mx, v);
  }
  smin[t] = mn; smax[t] = mx;
  __syncthreads();
  for (int s = 128; s > 0; s >>= 1) {
    if (t < s) { smin[t] = fminf(smin[t], smin[t + s]); smax[t] = fmaxf(smax[t], smax[t + s]); }
    __syncthreads();
  }
  if (t == 0) { mm[0] = smin[0]; mm[1] = smax[0]; }
}

__global__ void posadd_kernel(float* __restrict__ x, const float* __restrict__ ts,
                              const float* __restrict__ mm, const float* __restrict__ pos_emb,
                              const float* __restrict__ tscale) {
  const int e = blockIdx.x, c = threadIdx.x;
  float tmin = mm[0], tmax = mm[1];
  float posf = (tmax > tmin) ? (ts[e] - tmin) / (tmax - tmin) * 4999.0f : 0.0f;
  int pos = (int)posf;
  x[(size_t)e * HID + c] += tscale[0] * pos_emb[(size_t)pos * HID + c];
}

// ---------------------------------------------------------------------------
// stable counting sort by node id
// ---------------------------------------------------------------------------
__global__ void count_kernel(const int* __restrict__ key, int* __restrict__ deg) {
  int n = blockIdx.x * blockDim.x + threadIdx.x;
  if (n >= NNODES) return;
  int d = 0;
  for (int e = 0; e < E_EDGES; ++e) d += (key[e] == n);
  deg[n] = d;
}
__global__ void scan_kernel(const int* __restrict__ deg, int* __restrict__ off) {
  if (threadIdx.x == 0 && blockIdx.x == 0) {
    int r = 0;
    for (int n = 0; n < NNODES; ++n) { off[n] = r; r += deg[n]; }
  }
}
__global__ void scatter_kernel(const int* __restrict__ key, const int* __restrict__ off,
                               int* __restrict__ order) {
  int n = blockIdx.x * blockDim.x + threadIdx.x;
  if (n >= NNODES) return;
  int w = off[n];
  for (int e = 0; e < E_EDGES; ++e)
    if (key[e] == n) order[w++] = e;
}

// ---------------------------------------------------------------------------
// Per-segment LSTM scan. One wave per node; Wih part precomputed (xg).
// Lane owns hidden units 2*lane, 2*lane+1; hidden vector broadcast via LDS
// (one LDS read per k, reused across the lane's 8 gate rows; Whh stays
// L0-resident across steps).
// colbase = a*128 + d*64 selects the output slice of nf[n, 256].
// ---------------------------------------------------------------------------
__global__ void lstm_scan_kernel(const int* __restrict__ order, const int* __restrict__ off,
                                 const int* __restrict__ deg, const float* __restrict__ xg,
                                 const float* __restrict__ Whh, const float* __restrict__ bhh,
                                 const float* __restrict__ x, float* __restrict__ nf,
                                 int colbase, int reverse) {
  __shared__ float hsh[4][64];
  const int wave = threadIdx.x >> 5;
  const int lane = threadIdx.x & 31;
  const int node = blockIdx.x * 4 + wave;
  const int dg   = deg[node];
  const int base = off[node];
  float* h = hsh[wave];
  float* dst = nf + (size_t)node * 256 + colbase;

  if (dg == 0) {                           // no edges: zeros
    dst[lane * 2] = 0.0f;
    dst[lane * 2 + 1] = 0.0f;
    return;
  }
  if (dg == 1) {                           // deg==1: raw feature overrides LSTM
    int e = order[base];
    int srcc = (colbase & 127) + lane * 2; // d*64 + u
    dst[lane * 2]     = x[(size_t)e * HID + srcc];
    dst[lane * 2 + 1] = x[(size_t)e * HID + srcc + 1];
    return;
  }

  float c0 = 0.0f, c1 = 0.0f;
  h[lane * 2] = 0.0f;
  h[lane * 2 + 1] = 0.0f;
  for (int s = 0; s < dg; ++s) {
    int j = reverse ? (base + dg - 1 - s) : (base + s);
    int e = order[j];
    const float* gx = xg + (size_t)e * 256;
    float gates[8];                        // gates[t*2+p]: t=i,f,g,o ; p=unit parity
#pragma unroll
    for (int t = 0; t < 8; ++t) {
      int g = (t >> 1) * 64 + lane * 2 + (t & 1);
      gates[t] = gx[g] + bhh[g];
    }
    for (int kk = 0; kk < 64; ++kk) {
      float hk = h[kk];
#pragma unroll
      for (int t = 0; t < 8; ++t) {
        int g = (t >> 1) * 64 + lane * 2 + (t & 1);
        gates[t] += Whh[(size_t)g * 64 + kk] * hk;
      }
    }
#pragma unroll
    for (int p = 0; p < 2; ++p) {
      float ig = gates[p], fg = gates[2 + p], gg = gates[4 + p], og = gates[6 + p];
      float si = 1.0f / (1.0f + __expf(-ig));
      float sf = 1.0f / (1.0f + __expf(-fg));
      float so = 1.0f / (1.0f + __expf(-og));
      float cc = p ? c1 : c0;
      cc = sf * cc + si * tanhf(gg);
      if (p) c1 = cc; else c0 = cc;
      h[lane * 2 + p] = so * tanhf(cc);
    }
  }
  dst[lane * 2]     = h[lane * 2];
  dst[lane * 2 + 1] = h[lane * 2 + 1];
}

// ---------------------------------------------------------------------------
template<bool TRANSB, bool RELU, bool RESID>
static inline void gemm(const float* A, const float* B, const float* bias, const float* R,
                        float* C, int M, int Nn, int K, hipStream_t s) {
  dim3 grid((M / 16) * (Nn / 64));
  gemm_bf16_wmma<TRANSB, RELU, RESID><<<grid, 32, 0, s>>>(A, B, bias, R, C, M, Nn, K);
}

extern "C" void kernel_launch(void* const* d_in, const int* in_sizes, int n_in,
                              void* d_out, int out_size, void* d_ws, size_t ws_size,
                              hipStream_t stream) {
  (void)in_sizes; (void)n_in; (void)out_size; (void)ws_size;
  // -------- inputs (setup_inputs dict order, params flattened in insertion order)
  const int*   edge_index = (const int*)d_in[0];   // [2, E]
  const float* edge_attr  = (const float*)d_in[1]; // [E, 64]
  const float* edge_ts    = (const float*)d_in[2]; // [E]
  const float* proj_W = (const float*)d_in[4];
  const float* proj_b = (const float*)d_in[5];
  const float* pos_emb = (const float*)d_in[6];
  const float* tscale  = (const float*)d_in[7];
  const float* fusion_W = (const float*)d_in[56];
  const float* fusion_b = (const float*)d_in[57];

  // -------- workspace layout (floats)
  const size_t MEG = 1048576;
  float* f  = (float*)d_ws;
  float* x  = f;               // E*128
  float* t0 = f + 1 * MEG;     // E*128
  float* q  = f + 2 * MEG;
  float* k  = f + 3 * MEG;
  float* v  = f + 4 * MEG;
  float* o  = f + 5 * MEG;
  float* ff = f + 6 * MEG;     // E*512 (6..10 MEG)
  float* xg = f + 1 * MEG;     // 4 x E*256, reuses 1..9 MEG after transformer
  float* nf = f + 9 * MEG;     // 2048*256
  float* mm = f + 10 * MEG;    // 2 floats
  int*   ip = (int*)(f + 10 * MEG + 4);
  int* order_in  = ip;
  int* order_out = ip + E_EDGES;
  int* deg_in    = ip + 2 * E_EDGES;
  int* off_in    = deg_in + NNODES;
  int* deg_out   = off_in + NNODES;
  int* off_out   = deg_out + NNODES;

  // -------- 1) projection + temporal positional embedding
  gemm<false, false, false>(edge_attr, proj_W, proj_b, nullptr, x, E_EDGES, HID, 64, stream);
  minmax_kernel<<<1, 256, 0, stream>>>(edge_ts, mm);
  posadd_kernel<<<E_EDGES, HID, 0, stream>>>(x, edge_ts, mm, pos_emb, tscale);

  // -------- 2) transformer layers
  for (int L = 0; L < 2; ++L) {
    const int pb = 8 + L * 16;
    const float* Wq = (const float*)d_in[pb + 0];
    const float* bq = (const float*)d_in[pb + 1];
    const float* Wk = (const float*)d_in[pb + 2];
    const float* bk = (const float*)d_in[pb + 3];
    const float* Wv = (const float*)d_in[pb + 4];
    const float* bv = (const float*)d_in[pb + 5];
    const float* Wo = (const float*)d_in[pb + 6];
    const float* bo = (const float*)d_in[pb + 7];
    const float* W1 = (const float*)d_in[pb + 8];
    const float* b1 = (const float*)d_in[pb + 9];
    const float* W2 = (const float*)d_in[pb + 10];
    const float* b2 = (const float*)d_in[pb + 11];
    const float* g1 = (const float*)d_in[pb + 12];
    const float* be1 = (const float*)d_in[pb + 13];
    const float* g2 = (const float*)d_in[pb + 14];
    const float* be2 = (const float*)d_in[pb + 15];

    gemm<false, false, false>(x, Wq, bq, nullptr, q, E_EDGES, HID, HID, stream);
    gemm<false, false, false>(x, Wk, bk, nullptr, k, E_EDGES, HID, HID, stream);
    gemm<false, false, false>(x, Wv, bv, nullptr, v, E_EDGES, HID, HID, stream);
    flash_attn_kernel<<<dim3(E_EDGES / 16, HEADS), 32, 0, stream>>>(q, k, v, o);
    gemm<false, false, true>(o, Wo, bo, x, t0, E_EDGES, HID, HID, stream);
    ln_kernel<<<E_EDGES, HID, 0, stream>>>(t0, g1, be1, x);
    gemm<false, true, false>(x, W1, b1, nullptr, ff, E_EDGES, 4 * HID, HID, stream);
    gemm<false, false, true>(ff, W2, b2, x, t0, E_EDGES, HID, 4 * HID, stream);
    ln_kernel<<<E_EDGES, HID, 0, stream>>>(t0, g2, be2, x);
  }

  // -------- 3) stable counting sort by dst (in, edge_index[1]) / src (out, edge_index[0])
  const int* key_in  = edge_index + E_EDGES;  // row 1
  const int* key_out = edge_index;            // row 0
  count_kernel<<<NNODES / 256, 256, 0, stream>>>(key_in, deg_in);
  count_kernel<<<NNODES / 256, 256, 0, stream>>>(key_out, deg_out);
  scan_kernel<<<1, 64, 0, stream>>>(deg_in, off_in);
  scan_kernel<<<1, 64, 0, stream>>>(deg_out, off_out);
  scatter_kernel<<<NNODES / 256, 256, 0, stream>>>(key_in, off_in, order_in);
  scatter_kernel<<<NNODES / 256, 256, 0, stream>>>(key_out, off_out, order_out);

  // -------- 4) LSTM input-gate precompute: xg = x @ Wih^T + bih (per lstm, per dir)
  // in_lstm params at 40..47, out_lstm at 48..55:
  //   Wih_f, Whh_f, bih_f, bhh_f, Wih_b, Whh_b, bih_b, bhh_b
  float* xg_bufs[4] = { xg, xg + 2 * MEG, xg + 4 * MEG, xg + 6 * MEG };
  for (int a = 0; a < 2; ++a) {
    for (int d = 0; d < 2; ++d) {
      const float* Wih = (const float*)d_in[40 + a * 8 + d * 4 + 0];
      const float* bih = (const float*)d_in[40 + a * 8 + d * 4 + 2];
      gemm<true, false, false>(x, Wih, bih, nullptr, xg_bufs[a * 2 + d],
                               E_EDGES, 256, HID, stream);
    }
  }

  // -------- 5) per-node bidirectional LSTM scans (4 launches: a in/out, d fwd/bwd)
  for (int a = 0; a < 2; ++a) {
    const int* ordp = a ? order_out : order_in;
    const int* offp = a ? off_out : off_in;
    const int* degp = a ? deg_out : deg_in;
    for (int d = 0; d < 2; ++d) {
      const float* Whh = (const float*)d_in[40 + a * 8 + d * 4 + 1];
      const float* bhh = (const float*)d_in[40 + a * 8 + d * 4 + 3];
      lstm_scan_kernel<<<NNODES / 4, 128, 0, stream>>>(
          ordp, offp, degp, xg_bufs[a * 2 + d], Whh, bhh, x, nf,
          a * 128 + d * 64, d);
    }
  }

  // -------- 6) fusion: out = relu(nf @ fusion_W + fusion_b)
  gemm<false, true, false>(nf, fusion_W, fusion_b, nullptr, (float*)d_out,
                           NNODES, 256, 256, stream);
}
